// WindowAttention_34969623724201
// MI455X (gfx1250) — matrix operand order repaired and checked by reading, not compiled, not run on hardware
//
#include <hip/hip_runtime.h>
#include <hip/hip_bf16.h>

// ---------------------------------------------------------------------------
// Problem constants (from reference)
// ---------------------------------------------------------------------------
// B_ = 2048, N = 64, DIM = 192, C3 = 576, HEADS = 6, HEAD_DIM = 32, nW = 64
#define BATCH   2048
#define NTOK    64
#define DIMC    192
#define C3      576
#define HEADS   6
#define HDIM    32
#define ROWS    (BATCH * NTOK)          // 131072 tokens
#define SCALE_F 0.17677669529663687f    // 32^-0.5

typedef __attribute__((ext_vector_type(16))) __bf16 v16bf;
typedef __attribute__((ext_vector_type(8)))  float  v8f;
typedef __attribute__((ext_vector_type(4)))  unsigned int v4u;
typedef __attribute__((ext_vector_type(8)))  int v8i;
typedef __attribute__((ext_vector_type(4)))  int v4i;

union FragU { v16bf v; unsigned int u[8]; };

__device__ __forceinline__ unsigned short f2bf(float f) {
  unsigned int u = __builtin_bit_cast(unsigned int, f);
  u += 0x7FFFu + ((u >> 16) & 1u);            // round-to-nearest-even
  return (unsigned short)(u >> 16);
}
__device__ __forceinline__ float bf2f(unsigned short s) {
  unsigned int u = ((unsigned int)s) << 16;
  return __builtin_bit_cast(float, u);
}

#if defined(__gfx1250__) && __has_builtin(__builtin_amdgcn_tensor_load_to_lds) && \
    __has_builtin(__builtin_amdgcn_s_wait_tensorcnt)
#define USE_TDM 1
#else
#define USE_TDM 0
#endif

#if USE_TDM
// Generic LDS pointers keep the LDS byte offset in addr[31:0] (ISA 10.2).
__device__ __forceinline__ unsigned ldsOff(const void* p) {
  return (unsigned)(unsigned long long)p;
}
// 1-D TDM copy: n 2-byte elements, global -> LDS, with 1 dword of LDS padding
// inserted every 2^(pad_code+1) dwords (pad_code 3 = every 64B, 4 = every 128B).
// D# layout per CDNA5 ISA. This toolchain's
// builtin is the 6-operand form: (g0 x4, g1 x8, g2 x4, g3 x4, x8, cpol);
// groups 2/3 and the trailing group are zero for a <=2-D tensor.
__device__ __forceinline__ void tdm_load_1d(unsigned lds_byte_off, const void* gptr,
                                            unsigned n, unsigned pad_code) {
  const unsigned long long ga = (unsigned long long)gptr;
  const v4u g0 = { 1u,                                  // count=1, user mode
                   lds_byte_off,                        // lds_addr
                   (unsigned)ga,                        // global_addr[31:0]
                   ((unsigned)(ga >> 32) & 0x1FFFFFFu)  // global_addr[56:32]
                     | 0x80000000u };                   // type=2 ("image")
  const v8i g1 = { (int)((1u << 16)                     // data_size = 2 bytes
                       | (1u << 20)                     // pad_enable
                       | (pad_code << 22)),             // pad_interval (amount=0 -> 1 dword)
                   (int)(n << 16),                      // tensor_dim0[15:0] @ bits 63:48
                   (int)((n >> 16) | (1u << 16)),       // tensor_dim0 hi, tensor_dim1=1
                   (int)((n & 0xFFFFu) << 16),          // tile_dim0 @ bits 127:112
                   0,                                   // tile_dim1=0 (1-D), tile_dim2=0
                   (int)n,                              // tensor_dim0_stride low
                   0, 0 };
  const v4i z4 = { 0, 0, 0, 0 };
  const v8i z8 = { 0, 0, 0, 0, 0, 0, 0, 0 };
  __builtin_amdgcn_tensor_load_to_lds(g0, g1, z4, z4, z8, 0);
}
#endif

// ---------------------------------------------------------------------------
// Kernel 1 & 4: C[row,col] = sum_k A[row,k] * W[col,k] + bias[col]
// A: ROWS x 192 (f32 or bf16), W: Cn x 192 f32, out: ROWS x Cn (bf16 or f32).
// 64x64 tile per block, 256 threads = 8 waves, 2 output 16x16 tiles per wave,
// K staged in LDS as bf16, fragments built per CDNA5 16-bit WMMA layouts.
// ---------------------------------------------------------------------------
template<bool AF32, bool OF32>
__global__ __launch_bounds__(256) void gemm_bias_wmma(
    const void* __restrict__ Aptr, const float* __restrict__ W,
    const float* __restrict__ bias, void* __restrict__ Optr, int Cn)
{
  constexpr int K   = 192;
  constexpr int LDA = 198;                      // 99 dwords/row (odd) -> few bank conflicts
  __shared__ unsigned short As[64][LDA];
  __shared__ unsigned short Bs[64][LDA];

  const int ntb  = Cn >> 6;
  const int row0 = (blockIdx.x / ntb) << 6;
  const int col0 = (blockIdx.x % ntb) << 6;
  const int tid  = threadIdx.x;

  // ---- stage A tile (64 x 192) as bf16 ----
  if constexpr (AF32) {
    const float* A = (const float*)Aptr;
    #pragma unroll
    for (int i = 0; i < 12; ++i) {
      int e4 = i * 256 + tid;                   // float4 chunk id, 48 per row
      int rr = e4 / 48, cc = (e4 % 48) << 2;
      const float4 f = *(const float4*)(A + (size_t)(row0 + rr) * K + cc);
      As[rr][cc + 0] = f2bf(f.x); As[rr][cc + 1] = f2bf(f.y);
      As[rr][cc + 2] = f2bf(f.z); As[rr][cc + 3] = f2bf(f.w);
    }
  } else {
    const unsigned short* A = (const unsigned short*)Aptr;
    #pragma unroll
    for (int i = 0; i < 12; ++i) {
      int e4 = i * 256 + tid;
      int rr = e4 / 48, cc = (e4 % 48) << 2;
      const uint2 d = *(const uint2*)(A + (size_t)(row0 + rr) * K + cc);
      *(unsigned int*)&As[rr][cc]     = d.x;
      *(unsigned int*)&As[rr][cc + 2] = d.y;
    }
  }
  // ---- stage W tile (64 x 192) as bf16 ----
  #pragma unroll
  for (int i = 0; i < 12; ++i) {
    int e4 = i * 256 + tid;
    int rr = e4 / 48, cc = (e4 % 48) << 2;
    const float4 f = *(const float4*)(W + (size_t)(col0 + rr) * K + cc);
    Bs[rr][cc + 0] = f2bf(f.x); Bs[rr][cc + 1] = f2bf(f.y);
    Bs[rr][cc + 2] = f2bf(f.z); Bs[rr][cc + 3] = f2bf(f.w);
  }
  __syncthreads();

  const int wave = tid >> 5, lane = tid & 31;
  const int r = lane & 15, half = lane >> 4;
  const int tm0 = wave >> 2, tn = wave & 3, tm1 = tm0 + 2;   // tiles wave, wave+8
  const int ar0 = tm0 * 16 + r, ar1 = tm1 * 16 + r, bc = tn * 16 + r;

  v8f c0 = {0.f,0.f,0.f,0.f,0.f,0.f,0.f,0.f};
  v8f c1 = {0.f,0.f,0.f,0.f,0.f,0.f,0.f,0.f};

  #pragma unroll
  for (int k0 = 0; k0 < K; k0 += 32) {
    FragU a0, a1, fb;
    #pragma unroll
    for (int p = 0; p < 8; ++p) {
      // A layout: element e=2p -> K = (e/8)*16 + 8*half + e%8
      const int kka = k0 + ((p >> 2) << 4) + (half << 3) + ((p & 3) << 1);
      a0.u[p] = *(const unsigned int*)&As[ar0][kka];
      a1.u[p] = *(const unsigned int*)&As[ar1][kka];
      // B layout: element e=2p -> K = (e/8)*16 + 8*half + e%8
      const int kkb = k0 + ((p >> 2) << 4) + (half << 3) + ((p & 3) << 1);
      fb.u[p] = *(const unsigned int*)&Bs[bc][kkb];
    }
    c0 = __builtin_amdgcn_wmma_f32_16x16x32_bf16(false, a0.v, false, fb.v, (short)0, c0, false, false);
    c1 = __builtin_amdgcn_wmma_f32_16x16x32_bf16(false, a1.v, false, fb.v, (short)0, c1, false, false);
  }

  const float bv = bias[col0 + bc];
  if constexpr (OF32) {
    float* O = (float*)Optr;
    #pragma unroll
    for (int g = 0; g < 8; ++g) {
      O[(size_t)(row0 + tm0 * 16 + g + 8 * half) * Cn + col0 + bc] = c0[g] + bv;
      O[(size_t)(row0 + tm1 * 16 + g + 8 * half) * Cn + col0 + bc] = c1[g] + bv;
    }
  } else {
    unsigned short* O = (unsigned short*)Optr;
    #pragma unroll
    for (int g = 0; g < 8; ++g) {
      O[(size_t)(row0 + tm0 * 16 + g + 8 * half) * Cn + col0 + bc] = f2bf(c0[g] + bv);
      O[(size_t)(row0 + tm1 * 16 + g + 8 * half) * Cn + col0 + bc] = f2bf(c1[g] + bv);
    }
  }
}

// ---------------------------------------------------------------------------
// Kernel 2: depthwise 3x3 conv over the (B_, N) grid (zero pad), then route
// channel o = s*192 + h*32 + e into q (pre-scaled), k, and vT bf16 buffers.
// q,k: [b][h][n][e] ; vT: [b][h][e][n]  (v is only ever used as a B-matrix).
// ---------------------------------------------------------------------------
__global__ __launch_bounds__(256) void dwconv_route(
    const unsigned short* __restrict__ t, const float* __restrict__ dww,
    const float* __restrict__ dwb, unsigned short* __restrict__ qb,
    unsigned short* __restrict__ kb, unsigned short* __restrict__ vbT)
{
  const int idx = blockIdx.x * blockDim.x + threadIdx.x;   // exactly B_*N*C3 threads
  const int o = idx % C3;
  const int n = (idx / C3) & (NTOK - 1);
  const int b = idx / (C3 * NTOK);

  const float* wk = dww + o * 9;
  float acc = dwb[o];
  #pragma unroll
  for (int di = -1; di <= 1; ++di) {
    const int bb = b + di;
    if (bb < 0 || bb >= BATCH) continue;
    #pragma unroll
    for (int dj = -1; dj <= 1; ++dj) {
      const int nn = n + dj;
      if (nn < 0 || nn >= NTOK) continue;
      acc += bf2f(t[(size_t)(bb * NTOK + nn) * C3 + o]) * wk[(di + 1) * 3 + (dj + 1)];
    }
  }
  const int s = o / DIMC, rem = o % DIMC;
  const int h = rem >> 5, e = rem & 31;
  if (s == 0)      qb[(size_t)((b * HEADS + h) * NTOK + n) * HDIM + e] = f2bf(acc * SCALE_F);
  else if (s == 1) kb[(size_t)((b * HEADS + h) * NTOK + n) * HDIM + e] = f2bf(acc);
  else             vbT[(size_t)((b * HEADS + h) * HDIM + e) * NTOK + n] = f2bf(acc);
}

// ---------------------------------------------------------------------------
// Kernel 3: window attention per (b, h). 4 waves, wave w owns S rows [16w,16w+16).
// q/k/vT staged into LDS by the Tensor Data Mover (TDM descriptors with
// pad_enable reproducing the bank-conflict-avoiding strides 34 / 66 halfwords),
// completion on TENSORcnt. S = q kT (4 WMMA/wave, K=32=head_dim), +bias +mask,
// softmax via width-16 shfl_xor (a D-row spans 16 lanes of one half),
// P -> LDS -> A-fragments, O = P v (2x2 WMMA/wave, K=64), bf16 token-major out.
// ---------------------------------------------------------------------------
__global__ __launch_bounds__(128) void attn_wmma(
    const unsigned short* __restrict__ qb, const unsigned short* __restrict__ kb,
    const unsigned short* __restrict__ vbT, const float* __restrict__ mask,
    const float* __restrict__ rpb, const int* __restrict__ rel_idx,
    unsigned short* __restrict__ obuf)
{
  __shared__ unsigned short qs[64][34];        // 64B row + 1 dword pad
  __shared__ unsigned short ks[64][34];
  __shared__ unsigned short vs[32][66];        // vT: [e][n], 128B row + 1 dword pad
  __shared__ unsigned short Ps[4][16][66];     // per-wave P strip

  const int b = blockIdx.x / HEADS, h = blockIdx.x % HEADS;
  const int tid = threadIdx.x;
  const size_t base = (size_t)(b * HEADS + h) * (NTOK * HDIM);

#if USE_TDM
  if (tid < 32) {                              // wave 0 issues the DMA (EXEC ignored)
    tdm_load_1d(ldsOff(&qs[0][0]), qb + base, NTOK * HDIM, 3);   // pad every 64B
    tdm_load_1d(ldsOff(&ks[0][0]), kb + base, NTOK * HDIM, 3);
    tdm_load_1d(ldsOff(&vs[0][0]), vbT + base, NTOK * HDIM, 4);  // pad every 128B
    __builtin_amdgcn_s_wait_tensorcnt(0);
  }
  __syncthreads();
#else
  {
    const unsigned int* qsrc = (const unsigned int*)(qb + base);
    const unsigned int* ksrc = (const unsigned int*)(kb + base);
    const unsigned int* vsrc = (const unsigned int*)(vbT + base);
    #pragma unroll
    for (int i = 0; i < 8; ++i) {
      int idx = i * 128 + tid;                 // 1024 dwords per matrix
      int rr = idx >> 4, cc = (idx & 15) << 1;
      *(unsigned int*)&qs[rr][cc] = qsrc[idx];
      *(unsigned int*)&ks[rr][cc] = ksrc[idx];
      int vr = idx >> 5, vc = (idx & 31) << 1;
      *(unsigned int*)&vs[vr][vc] = vsrc[idx];
    }
    __syncthreads();
  }
#endif

  const int w = tid >> 5, lane = tid & 31;
  const int r = lane & 15, half = lane >> 4;

  // ---- S = q kT ----
  FragU fq;
  #pragma unroll
  for (int p = 0; p < 8; ++p) {
    const int kk = ((p >> 2) << 4) + (half << 3) + ((p & 3) << 1);
    fq.u[p] = *(const unsigned int*)&qs[w * 16 + r][kk];
  }
  v8f s[4];
  #pragma unroll
  for (int t = 0; t < 4; ++t) {
    FragU fk;
    #pragma unroll
    for (int p = 0; p < 8; ++p)
      fk.u[p] = *(const unsigned int*)&ks[t * 16 + r][((p >> 2) << 4) + (half << 3) + ((p & 3) << 1)];
    v8f z = {0.f,0.f,0.f,0.f,0.f,0.f,0.f,0.f};
    s[t] = __builtin_amdgcn_wmma_f32_16x16x32_bf16(false, fq.v, false, fk.v, (short)0, z, false, false);
  }

  // ---- bias + mask + softmax (row n lives on 16 lanes of one half) ----
  const int wb = (b & 63) << 12;               // window = b % nW, *N*N
  float sv[4][8];
  #pragma unroll
  for (int t = 0; t < 4; ++t) {
    const int mcol = t * 16 + r;
    #pragma unroll
    for (int g = 0; g < 8; ++g) {
      const int nrow = w * 16 + g + 8 * half;
      sv[t][g] = s[t][g] + rpb[rel_idx[nrow * 64 + mcol] * HEADS + h]
                         + mask[wb + nrow * 64 + mcol];
    }
  }
  #pragma unroll
  for (int g = 0; g < 8; ++g) {
    float mx = fmaxf(fmaxf(sv[0][g], sv[1][g]), fmaxf(sv[2][g], sv[3][g]));
    #pragma unroll
    for (int off = 8; off > 0; off >>= 1) mx = fmaxf(mx, __shfl_xor(mx, off, 16));
    float sum = 0.f;
    #pragma unroll
    for (int t = 0; t < 4; ++t) { sv[t][g] = __expf(sv[t][g] - mx); sum += sv[t][g]; }
    #pragma unroll
    for (int off = 8; off > 0; off >>= 1) sum += __shfl_xor(sum, off, 16);
    const float inv = 1.f / sum;
    const int prow = g + 8 * half;
    #pragma unroll
    for (int t = 0; t < 4; ++t) Ps[w][prow][t * 16 + r] = f2bf(sv[t][g] * inv);
  }

  // ---- O = P v ----
  #pragma unroll
  for (int tn = 0; tn < 2; ++tn) {
    v8f o = {0.f,0.f,0.f,0.f,0.f,0.f,0.f,0.f};
    #pragma unroll
    for (int kk0 = 0; kk0 < 64; kk0 += 32) {
      FragU fa, fv;
      #pragma unroll
      for (int p = 0; p < 8; ++p) {
        fa.u[p] = *(const unsigned int*)&Ps[w][r][kk0 + ((p >> 2) << 4) + (half << 3) + ((p & 3) << 1)];
        fv.u[p] = *(const unsigned int*)&vs[tn * 16 + r][kk0 + ((p >> 2) << 4) + (half << 3) + ((p & 3) << 1)];
      }
      o = __builtin_amdgcn_wmma_f32_16x16x32_bf16(false, fa.v, false, fv.v, (short)0, o, false, false);
    }
    #pragma unroll
    for (int g = 0; g < 8; ++g) {
      const int n = w * 16 + g + 8 * half;
      obuf[(size_t)(b * NTOK + n) * DIMC + h * HDIM + tn * 16 + r] = f2bf(o[g]);
    }
  }
}

// ---------------------------------------------------------------------------
// Launcher
// ---------------------------------------------------------------------------
extern "C" void kernel_launch(void* const* d_in, const int* in_sizes, int n_in,
                              void* d_out, int out_size, void* d_ws, size_t ws_size,
                              hipStream_t stream) {
  (void)in_sizes; (void)n_in; (void)out_size; (void)ws_size;
  const float* x      = (const float*)d_in[0];
  const float* mask   = (const float*)d_in[1];
  const float* qkv_w  = (const float*)d_in[2];
  const float* qkv_b  = (const float*)d_in[3];
  const float* dw_w   = (const float*)d_in[4];
  const float* dw_b   = (const float*)d_in[5];
  const float* rpb    = (const float*)d_in[6];
  const int*   rel    = (const int*)d_in[7];
  const float* proj_w = (const float*)d_in[8];
  const float* proj_b = (const float*)d_in[9];
  float* out = (float*)d_out;

  char* ws = (char*)d_ws;
  const size_t T_BYTES  = (size_t)ROWS * C3 * 2;            // 150,994,944
  const size_t QK_BYTES = (size_t)BATCH * HEADS * NTOK * HDIM * 2; // 50,331,648
  unsigned short* t    = (unsigned short*)ws;
  unsigned short* qbuf = (unsigned short*)(ws + T_BYTES);
  unsigned short* kbuf = (unsigned short*)(ws + T_BYTES + QK_BYTES);
  unsigned short* vbuf = (unsigned short*)(ws + T_BYTES + 2 * QK_BYTES);
  unsigned short* obuf = t;                                  // t dead after conv

  // 1) t = x @ qkv_w^T + qkv_b          (131072 x 576, K=192)
  gemm_bias_wmma<true,  false><<<(ROWS / 64) * (C3 / 64),   256, 0, stream>>>(x, qkv_w, qkv_b, t, C3);
  // 2) depthwise 3x3 over (B_, N) grid, route into q (scaled), k, vT
  dwconv_route<<<(BATCH * NTOK * C3) / 256, 256, 0, stream>>>(t, dw_w, dw_b, qbuf, kbuf, vbuf);
  // 3) per-(b,h) window attention -> obuf (token-major bf16)
  attn_wmma<<<BATCH * HEADS, 128, 0, stream>>>(qbuf, kbuf, vbuf, mask, rpb, rel, obuf);
  // 4) out = obuf @ proj_w^T + proj_b   (131072 x 192, K=192), f32 output
  gemm_bias_wmma<false, true><<<(ROWS / 64) * (DIMC / 64),  256, 0, stream>>>(obuf, proj_w, proj_b, out, DIMC);
}
